// PrometheusV4Transformer_77799037599850
// MI455X (gfx1250) — matrix operand order repaired and checked by reading, not compiled, not run on hardware
//
#include <hip/hip_runtime.h>
#include <hip/hip_bf16.h>

typedef _Float16 v8h  __attribute__((ext_vector_type(8)));
typedef _Float16 v16h __attribute__((ext_vector_type(16)));
typedef float    v8f  __attribute__((ext_vector_type(8)));

#define BATCH 16
#define GDIM 30
#define SEQ 900
#define SVT 960           // padded S stride for transposed V (16B-aligned rows)
#define DMODEL 256
#define NLAYER 6
#define NHEAD 8
#define DKH 32
#define MTOT (BATCH*SEQ)  // 14400
#define NOPS 16
#define STEPS 8

__device__ __forceinline__ v16h cat8(v8h a, v8h b) {
  return __builtin_shufflevector(a, b, 0,1,2,3,4,5,6,7,8,9,10,11,12,13,14,15);
}
__device__ __forceinline__ v8f wmma16x16x32(v16h a, v16h b, v8f c) {
  return __builtin_amdgcn_wmma_f32_16x16x32_f16(false, a, false, b, (short)0, c, false, false);
}
__device__ __forceinline__ float gelu_exact(float x) {
  return 0.5f * x * (1.0f + erff(x * 0.70710678118654752f));
}

// ---------------- store modes for the generic WMMA GEMM ----------------
#define ST_F16       0   // f16 [M,N]
#define ST_F32_RES   1   // f32 [M,N] = acc + bias + resid
#define ST_F16_GELU  2   // f16 gelu(acc+bias)
#define ST_F16_SCALE 3   // f16 (acc+bias)*scale      (Q with 1/sqrt(DK))
#define ST_VT        4   // f16 scatter to V^T [B,NH,DKH,SVT]

// C = A[M,K](f16,row-major) x Bp(packed f16 [K/32][N][32]) + bias, epilogue by mode.
// Each wave computes a 32x32 tile via 4 v_wmma_f32_16x16x32_f16 per K-chunk.
__global__ __launch_bounds__(256)
void wmma_gemm(const _Float16* __restrict__ A, const _Float16* __restrict__ Bp,
               const float* __restrict__ bias, void* __restrict__ out,
               const float* __restrict__ resid,
               int M, int N, int K, int mode, float scale)
{
  const int lane = threadIdx.x & 31;
  const int wave = threadIdx.x >> 5;
  const int half = lane >> 4;
  const int l15  = lane & 15;
  const int tN = blockIdx.x;
  const int tM = blockIdx.y * 8 + wave;
  if (tM * 32 >= M) return;
  const int row0 = tM * 32, col0 = tN * 32;
  const int koff = half * 8;

  v8f acc[2][2] = {};
  for (int k0 = 0; k0 < K; k0 += 32) {
    v16h af[2], bf[2];
#pragma unroll
    for (int mi = 0; mi < 2; ++mi) {
      const _Float16* ap = A + (size_t)(row0 + mi*16 + l15) * K + k0;
      af[mi] = cat8(*(const v8h*)(ap + koff), *(const v8h*)(ap + 16 + koff));
    }
#pragma unroll
    for (int ni = 0; ni < 2; ++ni) {
      const _Float16* bp = Bp + (size_t)(k0 >> 5) * N * 32
                              + (size_t)(col0 + ni*16 + l15) * 32 + half * 8;
      bf[ni] = cat8(*(const v8h*)bp, *(const v8h*)(bp + 16));
    }
#pragma unroll
    for (int mi = 0; mi < 2; ++mi)
#pragma unroll
      for (int ni = 0; ni < 2; ++ni)
        acc[mi][ni] = wmma16x16x32(af[mi], bf[ni], acc[mi][ni]);
  }

#pragma unroll
  for (int mi = 0; mi < 2; ++mi) {
#pragma unroll
    for (int ni = 0; ni < 2; ++ni) {
      const int n = col0 + ni*16 + l15;
      const float bs = bias[n];
#pragma unroll
      for (int r = 0; r < 8; ++r) {
        const int m = row0 + mi*16 + (half ? r + 8 : r);
        float v = acc[mi][ni][r] + bs;
        if (mode == ST_F32_RES) {
          ((float*)out)[(size_t)m * N + n] = v + resid[(size_t)m * N + n];
        } else if (mode == ST_VT) {
          const int bb = m / SEQ, s = m - bb * SEQ;
          const int hh = n >> 5, dk = n & 31;
          ((_Float16*)out)[(((size_t)bb*NHEAD + hh)*DKH + dk)*SVT + s] = (_Float16)v;
        } else {
          if (mode == ST_F16_GELU)       v = gelu_exact(v);
          else if (mode == ST_F16_SCALE) v *= scale;
          ((_Float16*)out)[(size_t)m * N + n] = (_Float16)v;
        }
      }
    }
  }
}

// ---------------- flash attention: 1 wave per (qtile, head, batch) ----------------
__global__ __launch_bounds__(32)
void attn_kernel(const _Float16* __restrict__ Q, const _Float16* __restrict__ Km,
                 const _Float16* __restrict__ Vt, const float* __restrict__ sb,
                 _Float16* __restrict__ Ao)
{
  const int lane = threadIdx.x & 31, half = lane >> 4, l15 = lane & 15;
  const int qt = blockIdx.x, h = blockIdx.y, b = blockIdx.z;
  __shared__ _Float16 pt[16 * 32];

  const int qrow = qt * 16 + l15;
  const int qc = qrow < SEQ ? qrow : SEQ - 1;
  const _Float16* qp = Q + ((size_t)b * SEQ + qc) * DMODEL + h * DKH;
  const v16h aq = cat8(*(const v8h*)(qp + half*8), *(const v8h*)(qp + 16 + half*8));

  v8f o0 = {}, o1 = {};
  float mrow[8], lrow[8];
  int rq[8];
#pragma unroll
  for (int r = 0; r < 8; ++r) {
    mrow[r] = -1e30f; lrow[r] = 0.f;
    int m = qt * 16 + (half ? r + 8 : r);
    rq[r] = m < SEQ ? m : SEQ - 1;
  }

  for (int jb = 0; jb < 29; ++jb) {
    const int jbase = jb * 32;
    const int key0 = jbase + l15, key1 = jbase + 16 + l15;
    const int k0c = key0 < SEQ ? key0 : SEQ - 1;
    const int k1c = key1 < SEQ ? key1 : SEQ - 1;
    const _Float16* kp0 = Km + ((size_t)b * SEQ + k0c) * DMODEL + h * DKH + half * 8;
    const _Float16* kp1 = Km + ((size_t)b * SEQ + k1c) * DMODEL + h * DKH + half * 8;
    const v16h kb0 = cat8(*(const v8h*)kp0, *(const v8h*)(kp0 + 16));
    const v16h kb1 = cat8(*(const v8h*)kp1, *(const v8h*)(kp1 + 16));
    v8f s0 = {}, s1 = {};
    s0 = wmma16x16x32(aq, kb0, s0);
    s1 = wmma16x16x32(aq, kb1, s1);

    const int cr0 = k0c / GDIM, cc0 = k0c % GDIM;
    const int cr1 = k1c / GDIM, cc1 = k1c % GDIM;
    const bool v0 = key0 < SEQ, v1 = key1 < SEQ;
#pragma unroll
    for (int r = 0; r < 8; ++r) {
      const int rr = rq[r] / GDIM, rc = rq[r] % GDIM;
      int d0 = (rr > cr0 ? rr - cr0 : cr0 - rr) + (rc > cc0 ? rc - cc0 : cc0 - rc);
      int d1 = (rr > cr1 ? rr - cr1 : cr1 - rr) + (rc > cc1 ? rc - cc1 : cc1 - rc);
      float x0 = v0 ? (s0[r] + sb[d0 * 59]) : -1e30f;
      float x1 = v1 ? (s1[r] + sb[d1 * 59]) : -1e30f;
      float tmax = fmaxf(x0, x1);
      tmax = fmaxf(tmax, __shfl_xor(tmax, 1, 32));
      tmax = fmaxf(tmax, __shfl_xor(tmax, 2, 32));
      tmax = fmaxf(tmax, __shfl_xor(tmax, 4, 32));
      tmax = fmaxf(tmax, __shfl_xor(tmax, 8, 32));
      const float mnew = fmaxf(mrow[r], tmax);
      const float sc = __expf(mrow[r] - mnew);
      o0[r] = o0[r] * sc; o1[r] = o1[r] * sc; lrow[r] *= sc; mrow[r] = mnew;
      const float p0 = __expf(x0 - mnew), p1 = __expf(x1 - mnew);
      float ps = p0 + p1;
      ps += __shfl_xor(ps, 1, 32); ps += __shfl_xor(ps, 2, 32);
      ps += __shfl_xor(ps, 4, 32); ps += __shfl_xor(ps, 8, 32);
      lrow[r] += ps;
      const int rw = half ? r + 8 : r;
      pt[rw * 32 + l15]      = (_Float16)p0;
      pt[rw * 32 + 16 + l15] = (_Float16)p1;
    }
    __syncthreads();
    const _Float16* pr = &pt[l15 * 32];
    const v16h pa = cat8(*(const v8h*)(pr + half*8), *(const v8h*)(pr + 16 + half*8));
    const _Float16* vp0 = Vt + (((size_t)b*NHEAD + h)*DKH + l15     )*SVT + jbase + half*8;
    const _Float16* vp1 = Vt + (((size_t)b*NHEAD + h)*DKH + 16 + l15)*SVT + jbase + half*8;
    const v16h vb0 = cat8(*(const v8h*)vp0, *(const v8h*)(vp0 + 16));
    const v16h vb1 = cat8(*(const v8h*)vp1, *(const v8h*)(vp1 + 16));
    o0 = wmma16x16x32(pa, vb0, o0);
    o1 = wmma16x16x32(pa, vb1, o1);
    __syncthreads();
  }

#pragma unroll
  for (int r = 0; r < 8; ++r) {
    const int m = qt * 16 + (half ? r + 8 : r);
    if (m < SEQ) {
      const float iv = 1.0f / lrow[r];
      const size_t base = ((size_t)b * SEQ + m) * DMODEL + h * DKH;
      Ao[base + l15]      = (_Float16)(o0[r] * iv);
      Ao[base + 16 + l15] = (_Float16)(o1[r] * iv);
    }
  }
}

// ---------------- supporting elementwise / small kernels ----------------
__global__ void pack_w_kernel(const float* __restrict__ W, _Float16* __restrict__ Bp,
                              int K, int N) {
  size_t id = (size_t)blockIdx.x * 256 + threadIdx.x;
  if (id >= (size_t)K * N) return;
  int n = (int)(id % N); int k = (int)(id / N);
  Bp[(size_t)(k >> 5) * N * 32 + (size_t)n * 32 + (k & 31)] = (_Float16)W[id];
}

__global__ void zero_f16_kernel(_Float16* p, size_t n) {
  size_t id = (size_t)blockIdx.x * 256 + threadIdx.x;
  if (id < n) p[id] = (_Float16)0.0f;
}

__global__ void embed_kernel(const float* __restrict__ grid, const float* __restrict__ W,
                             const float* __restrict__ bias, const float* __restrict__ pe,
                             float* __restrict__ xs, _Float16* __restrict__ xsh) {
  size_t id = (size_t)blockIdx.x * 256 + threadIdx.x;
  if (id >= (size_t)MTOT * DMODEL) return;
  int d = (int)(id % DMODEL); size_t t = id / DMODEL;
  int s = (int)(t % SEQ); int b = (int)(t / SEQ);
  float a = bias[d] + pe[(size_t)s * DMODEL + d];
#pragma unroll
  for (int c = 0; c < 10; ++c)
    a += grid[((size_t)b * 10 + c) * SEQ + s] * W[c * DMODEL + d];
  xs[id] = a; xsh[id] = (_Float16)a;
}

__global__ __launch_bounds__(256)
void ln_kernel(const float* __restrict__ x, const float* __restrict__ g,
               const float* __restrict__ bb, float* __restrict__ of32,
               _Float16* __restrict__ of16) {
  __shared__ float red[256];
  const int row = blockIdx.x, tid = threadIdx.x;
  const float v = x[(size_t)row * DMODEL + tid];
  red[tid] = v; __syncthreads();
  for (int s = 128; s > 0; s >>= 1) { if (tid < s) red[tid] += red[tid + s]; __syncthreads(); }
  const float mean = red[0] * (1.0f / DMODEL); __syncthreads();
  const float d = v - mean;
  red[tid] = d * d; __syncthreads();
  for (int s = 128; s > 0; s >>= 1) { if (tid < s) red[tid] += red[tid + s]; __syncthreads(); }
  const float var = red[0] * (1.0f / DMODEL);
  const float y = d * rsqrtf(var + 1e-5f) * g[tid] + bb[tid];
  of32[(size_t)row * DMODEL + tid] = y;
  of16[(size_t)row * DMODEL + tid] = (_Float16)y;
}

__global__ void mean_kernel(const float* __restrict__ xs, float* __restrict__ gf) {
  int id = blockIdx.x * 256 + threadIdx.x;
  if (id >= BATCH * DMODEL) return;
  int d = id & 255, b = id >> 8;
  float s = 0.f;
  for (int t = 0; t < SEQ; ++t) s += xs[((size_t)b * SEQ + t) * DMODEL + d];
  gf[id] = s * (1.0f / SEQ);
}

__global__ void pg1_kernel(const float* __restrict__ gf, const float* __restrict__ W,
                           const float* __restrict__ bi, float* __restrict__ out) {
  int id = blockIdx.x * 256 + threadIdx.x;
  if (id >= BATCH * 512) return;
  int j = id % 512, b = id / 512;
  float a = bi[j];
  for (int k = 0; k < DMODEL; ++k) a += gf[b * DMODEL + k] * W[k * 512 + j];
  out[id] = gelu_exact(a);
}

__global__ void pg2_kernel(const float* __restrict__ hh, const float* __restrict__ W,
                           const float* __restrict__ bi, float* __restrict__ out) {
  int id = blockIdx.x * 256 + threadIdx.x;
  if (id >= BATCH * 128) return;
  int j = id % 128, b = id / 128;
  float a = bi[j];
  for (int k = 0; k < 512; ++k) a += hh[b * 512 + k] * W[k * 128 + j];
  out[id] = a;
}

__global__ void softmax_probs_kernel(const float* __restrict__ logits, float* __restrict__ probs) {
  int id = blockIdx.x * 128 + threadIdx.x;
  if (id >= BATCH * STEPS) return;
  int b = id / STEPS, st = id % STEPS;
  const float* L = logits + b * (STEPS * NOPS) + st * NOPS;
  float mx = -1e30f;
  for (int o = 0; o < NOPS; ++o) mx = fmaxf(mx, L[o]);
  float sum = 0.f, e[NOPS];
  for (int o = 0; o < NOPS; ++o) { e[o] = __expf(L[o] - mx); sum += e[o]; }
  float inv = 1.0f / sum;
  float* P = probs + b * (STEPS * NOPS) + st * NOPS;
  for (int o = 0; o < NOPS; ++o) P[o] = e[o] * inv;
}

__global__ void copy_f32_kernel(const float* __restrict__ s, float* __restrict__ d, size_t n) {
  size_t id = (size_t)blockIdx.x * 256 + threadIdx.x;
  if (id < n) d[id] = s[id];
}

__global__ void conv1_kernel(const float* __restrict__ grid, const float* __restrict__ cw1,
                             const float* __restrict__ cb1, float* __restrict__ h1) {
  size_t id = (size_t)blockIdx.x * 256 + threadIdx.x;
  if (id >= (size_t)NOPS * BATCH * 32 * SEQ) return;
  int p = (int)(id % SEQ); size_t t = id / SEQ;
  int co = (int)(t % 32); t /= 32;
  int b = (int)(t % BATCH); int o = (int)(t / BATCH);
  int h = p / GDIM, w = p % GDIM;
  float a = cb1[o * 32 + co];
  for (int ci = 0; ci < 10; ++ci) {
    const float* gp = grid + ((size_t)b * 10 + ci) * SEQ;
    const float* wp = cw1 + (((size_t)o * 32 + co) * 10 + ci) * 9;
#pragma unroll
    for (int kh = 0; kh < 3; ++kh) {
      int ih = h + kh - 1;
      if (ih < 0 || ih >= GDIM) continue;
#pragma unroll
      for (int kw = 0; kw < 3; ++kw) {
        int iw = w + kw - 1;
        if (iw < 0 || iw >= GDIM) continue;
        a += gp[ih * GDIM + iw] * wp[kh * 3 + kw];
      }
    }
  }
  h1[id] = fmaxf(a, 0.f);
}

__global__ void conv2_kernel(const float* __restrict__ h1, const float* __restrict__ cw2,
                             const float* __restrict__ cb2, float* __restrict__ allr) {
  size_t id = (size_t)blockIdx.x * 256 + threadIdx.x;
  if (id >= (size_t)NOPS * BATCH * 10 * SEQ) return;
  int p = (int)(id % SEQ); size_t t = id / SEQ;
  int c = (int)(t % 10); t /= 10;
  int b = (int)(t % BATCH); int o = (int)(t / BATCH);
  float a = cb2[o * 10 + c];
  const float* hp = h1 + (((size_t)o * BATCH + b) * 32) * SEQ + p;
  const float* wp = cw2 + ((size_t)o * 10 + c) * 32;
#pragma unroll
  for (int ci = 0; ci < 32; ++ci) a += hp[(size_t)ci * SEQ] * wp[ci];
  allr[id] = tanhf(a);
}

__global__ void mix_kernel(const float* __restrict__ allr, const float* __restrict__ probs,
                           int step, float* __restrict__ gout) {
  int id = blockIdx.x * 256 + threadIdx.x;
  if (id >= BATCH * 10 * SEQ) return;
  int p = id % SEQ; int t = id / SEQ;
  int c = t % 10; int b = t / 10;
  float a = 0.f;
#pragma unroll
  for (int o = 0; o < NOPS; ++o)
    a += probs[b * (STEPS * NOPS) + step * NOPS + o] *
         allr[(((size_t)o * BATCH + b) * 10 + c) * SEQ + p];
  gout[id] = a;
}

__global__ void final_kernel(const _Float16* __restrict__ op1, const float* __restrict__ oW2,
                             const float* __restrict__ ob2, const float* __restrict__ grid,
                             float* __restrict__ out) {
  int id = blockIdx.x * 256 + threadIdx.x;
  if (id >= BATCH * 10 * SEQ) return;
  int p = id % SEQ; int t = id / SEQ;
  int c = t % 10; int b = t / 10;
  const size_t m = (size_t)b * SEQ + p;
  float a = ob2[c];
  for (int k = 0; k < 128; ++k) a += (float)op1[m * 128 + k] * oW2[k * 10 + c];
  out[id] = 0.7f * a + 0.3f * grid[((size_t)b * 10 + c) * SEQ + p];
}

// ---------------- host-side orchestration ----------------
extern "C" void kernel_launch(void* const* d_in, const int* in_sizes, int n_in,
                              void* d_out, int out_size, void* d_ws, size_t ws_size,
                              hipStream_t stream) {
  const float* in_grid = (const float*)d_in[0];
  const float* emb_W = (const float*)d_in[1];
  const float* emb_b = (const float*)d_in[2];
  const float* pe    = (const float*)d_in[3];
  const float* Wq = (const float*)d_in[4];  const float* bq = (const float*)d_in[5];
  const float* Wk = (const float*)d_in[6];  const float* bk = (const float*)d_in[7];
  const float* Wv = (const float*)d_in[8];  const float* bv = (const float*)d_in[9];
  const float* Wo = (const float*)d_in[10]; const float* bo = (const float*)d_in[11];
  const float* sbias = (const float*)d_in[12];
  const float* ln1_g = (const float*)d_in[13]; const float* ln1_b = (const float*)d_in[14];
  const float* W1 = (const float*)d_in[15]; const float* b1 = (const float*)d_in[16];
  const float* W2 = (const float*)d_in[17]; const float* b2 = (const float*)d_in[18];
  const float* ln2_g = (const float*)d_in[19]; const float* ln2_b = (const float*)d_in[20];
  const float* pgW1 = (const float*)d_in[21]; const float* pgb1 = (const float*)d_in[22];
  const float* pgW2 = (const float*)d_in[23]; const float* pgb2 = (const float*)d_in[24];
  const float* cw1 = (const float*)d_in[25]; const float* cb1 = (const float*)d_in[26];
  const float* cw2 = (const float*)d_in[27]; const float* cb2 = (const float*)d_in[28];
  const float* oW1 = (const float*)d_in[29]; const float* ob1 = (const float*)d_in[30];
  const float* oW2 = (const float*)d_in[31]; const float* ob2 = (const float*)d_in[32];

  char* ws = (char*)d_ws;
  size_t off = 0;
  auto alloc = [&](size_t bytes) -> void* {
    void* p = ws + off; off += (bytes + 255) & ~(size_t)255; return p;
  };

  float*     xs_f32  = (float*)    alloc((size_t)MTOT * DMODEL * 4);
  float*     tmp_f32 = (float*)    alloc((size_t)MTOT * DMODEL * 4);
  float*     h_f32   = (float*)    alloc((size_t)MTOT * DMODEL * 4);
  _Float16*  xs_f16  = (_Float16*) alloc((size_t)MTOT * DMODEL * 2);
  _Float16*  h_f16   = (_Float16*) alloc((size_t)MTOT * DMODEL * 2);
  _Float16*  q_f16   = (_Float16*) alloc((size_t)MTOT * DMODEL * 2);
  _Float16*  k_f16   = (_Float16*) alloc((size_t)MTOT * DMODEL * 2);
  _Float16*  a_f16   = (_Float16*) alloc((size_t)MTOT * DMODEL * 2);
  const size_t vt_elems = (size_t)BATCH * NHEAD * DKH * SVT;
  _Float16*  vT_f16  = (_Float16*) alloc(vt_elems * 2 + 4096);
  _Float16*  ff1_f16 = (_Float16*) alloc((size_t)MTOT * 1024 * 2);   // aliased as conv h1 (f32)
  _Float16*  op1_f16 = (_Float16*) alloc((size_t)MTOT * 128 * 2);
  _Float16*  Wqp = (_Float16*) alloc((size_t)NLAYER * DMODEL * DMODEL * 2);
  _Float16*  Wkp = (_Float16*) alloc((size_t)NLAYER * DMODEL * DMODEL * 2);
  _Float16*  Wvp = (_Float16*) alloc((size_t)NLAYER * DMODEL * DMODEL * 2);
  _Float16*  Wop = (_Float16*) alloc((size_t)NLAYER * DMODEL * DMODEL * 2);
  _Float16*  W1p = (_Float16*) alloc((size_t)NLAYER * DMODEL * 1024 * 2);
  _Float16*  W2p = (_Float16*) alloc((size_t)NLAYER * 1024 * DMODEL * 2);
  _Float16*  oW1p = (_Float16*) alloc((size_t)DMODEL * 128 * 2);
  float* gfeat  = (float*)alloc(BATCH * DMODEL * 4);
  float* pgh    = (float*)alloc(BATCH * 512 * 4);
  float* logits = (float*)alloc(BATCH * 128 * 4);
  float* probs  = (float*)alloc(BATCH * STEPS * NOPS * 4);
  float* gridA  = (float*)alloc((size_t)BATCH * 10 * SEQ * 4);
  float* gridB  = (float*)alloc((size_t)BATCH * 10 * SEQ * 4);
  float* h1   = (float*)ff1_f16;  // 29.5 MB, conv phase only
  float* allr = (float*)tmp_f32;  // 9.2 MB fits in 14.7 MB, conv phase only

  auto blk = [](size_t n) { return (unsigned)((n + 255) / 256); };

  // ---- pack weights to WMMA-friendly f16 layout (every launch; deterministic) ----
  for (int l = 0; l < NLAYER; ++l) {
    hipLaunchKernelGGL(pack_w_kernel, dim3(blk(65536)), dim3(256), 0, stream,
                       Wq + (size_t)l*65536, Wqp + (size_t)l*65536, 256, 256);
    hipLaunchKernelGGL(pack_w_kernel, dim3(blk(65536)), dim3(256), 0, stream,
                       Wk + (size_t)l*65536, Wkp + (size_t)l*65536, 256, 256);
    hipLaunchKernelGGL(pack_w_kernel, dim3(blk(65536)), dim3(256), 0, stream,
                       Wv + (size_t)l*65536, Wvp + (size_t)l*65536, 256, 256);
    hipLaunchKernelGGL(pack_w_kernel, dim3(blk(65536)), dim3(256), 0, stream,
                       Wo + (size_t)l*65536, Wop + (size_t)l*65536, 256, 256);
    hipLaunchKernelGGL(pack_w_kernel, dim3(blk(262144)), dim3(256), 0, stream,
                       W1 + (size_t)l*262144, W1p + (size_t)l*262144, 256, 1024);
    hipLaunchKernelGGL(pack_w_kernel, dim3(blk(262144)), dim3(256), 0, stream,
                       W2 + (size_t)l*262144, W2p + (size_t)l*262144, 1024, 256);
  }
  hipLaunchKernelGGL(pack_w_kernel, dim3(blk(32768)), dim3(256), 0, stream,
                     oW1, oW1p, 256, 128);
  hipLaunchKernelGGL(zero_f16_kernel, dim3(blk(vt_elems + 2048)), dim3(256), 0, stream,
                     vT_f16, vt_elems + 2048);

  // ---- embedding + positional encoding ----
  hipLaunchKernelGGL(embed_kernel, dim3(blk((size_t)MTOT * DMODEL)), dim3(256), 0, stream,
                     in_grid, emb_W, emb_b, pe, xs_f32, xs_f16);

  const dim3 gB(1, (MTOT / 32 + 7) / 8), tB(256);
  const float qscale = 0.17677669529663687f; // 1/sqrt(32)

  // ---- transformer layers ----
  for (int l = 0; l < NLAYER; ++l) {
    const size_t w256 = (size_t)l * 65536, wff = (size_t)l * 262144;
    hipLaunchKernelGGL(wmma_gemm, dim3(DMODEL/32, gB.y), tB, 0, stream,
        xs_f16, Wqp + w256, bq + l*DMODEL, (void*)q_f16, (const float*)nullptr,
        MTOT, DMODEL, DMODEL, ST_F16_SCALE, qscale);
    hipLaunchKernelGGL(wmma_gemm, dim3(DMODEL/32, gB.y), tB, 0, stream,
        xs_f16, Wkp + w256, bk + l*DMODEL, (void*)k_f16, (const float*)nullptr,
        MTOT, DMODEL, DMODEL, ST_F16, 1.0f);
    hipLaunchKernelGGL(wmma_gemm, dim3(DMODEL/32, gB.y), tB, 0, stream,
        xs_f16, Wvp + w256, bv + l*DMODEL, (void*)vT_f16, (const float*)nullptr,
        MTOT, DMODEL, DMODEL, ST_VT, 1.0f);
    hipLaunchKernelGGL(attn_kernel, dim3(57, NHEAD, BATCH), dim3(32), 0, stream,
        q_f16, k_f16, vT_f16, sbias + (size_t)l*3481, a_f16);
    hipLaunchKernelGGL(wmma_gemm, dim3(DMODEL/32, gB.y), tB, 0, stream,
        a_f16, Wop + w256, bo + l*DMODEL, (void*)tmp_f32, xs_f32,
        MTOT, DMODEL, DMODEL, ST_F32_RES, 1.0f);
    hipLaunchKernelGGL(ln_kernel, dim3(MTOT), dim3(256), 0, stream,
        tmp_f32, ln1_g + l*DMODEL, ln1_b + l*DMODEL, h_f32, h_f16);
    hipLaunchKernelGGL(wmma_gemm, dim3(1024/32, gB.y), tB, 0, stream,
        h_f16, W1p + wff, b1 + l*1024, (void*)ff1_f16, (const float*)nullptr,
        MTOT, 1024, DMODEL, ST_F16_GELU, 1.0f);
    hipLaunchKernelGGL(wmma_gemm, dim3(DMODEL/32, gB.y), tB, 0, stream,
        ff1_f16, W2p + wff, b2 + l*DMODEL, (void*)tmp_f32, h_f32,
        MTOT, DMODEL, 1024, ST_F32_RES, 1.0f);
    hipLaunchKernelGGL(ln_kernel, dim3(MTOT), dim3(256), 0, stream,
        tmp_f32, ln2_g + l*DMODEL, ln2_b + l*DMODEL, xs_f32, xs_f16);
  }

  // ---- program-synthesis head ----
  hipLaunchKernelGGL(mean_kernel, dim3(blk(BATCH*DMODEL)), dim3(256), 0, stream, xs_f32, gfeat);
  hipLaunchKernelGGL(pg1_kernel, dim3(blk(BATCH*512)), dim3(256), 0, stream, gfeat, pgW1, pgb1, pgh);
  hipLaunchKernelGGL(pg2_kernel, dim3(blk(BATCH*128)), dim3(256), 0, stream, pgh, pgW2, pgb2, logits);
  hipLaunchKernelGGL(softmax_probs_kernel, dim3(1), dim3(128), 0, stream, logits, probs);

  // ---- output projection hidden (uses final feats) ----
  hipLaunchKernelGGL(wmma_gemm, dim3(128/32, gB.y), tB, 0, stream,
      xs_f16, oW1p, ob1, (void*)op1_f16, (const float*)nullptr,
      MTOT, 128, DMODEL, ST_F16_GELU, 1.0f);

  // ---- conv expert program (reuses ff1/tmp scratch) ----
  hipLaunchKernelGGL(copy_f32_kernel, dim3(blk((size_t)BATCH*10*SEQ)), dim3(256), 0, stream,
                     in_grid, gridA, (size_t)BATCH*10*SEQ);
  for (int st = 0; st < STEPS; ++st) {
    float* src = (st & 1) ? gridB : gridA;
    float* dst = (st & 1) ? gridA : gridB;
    hipLaunchKernelGGL(conv1_kernel, dim3(blk((size_t)NOPS*BATCH*32*SEQ)), dim3(256), 0, stream,
                       src, cw1, cb1, h1);
    hipLaunchKernelGGL(conv2_kernel, dim3(blk((size_t)NOPS*BATCH*10*SEQ)), dim3(256), 0, stream,
                       h1, cw2, cb2, allr);
    hipLaunchKernelGGL(mix_kernel, dim3(blk(BATCH*10*SEQ)), dim3(256), 0, stream,
                       allr, probs, st, dst);
  }
  float* gfinal = gridA; // STEPS=8: last write (st=7) lands in gridA

  // ---- final blend ----
  hipLaunchKernelGGL(final_kernel, dim3(blk(BATCH*10*SEQ)), dim3(256), 0, stream,
                     op1_f16, oW2, ob2, gfinal, (float*)d_out);
}
